// OPMean_84430467105270
// MI455X (gfx1250) — compile-verified, matrix operand order checked by reading, not run on hardware
//
#include <hip/hip_runtime.h>
#include <hip/hip_bf16.h>
#include <stdint.h>

typedef __attribute__((ext_vector_type(16))) __bf16 v16bf;
typedef __attribute__((ext_vector_type(8)))  float  v8f;

#define CM   256
#define HM   128   // s
#define WM   256   // i / j
#define CSM  32
#define CZ   128
#define NTOT (CM * HM * WM)  // 8388608

static __device__ __forceinline__ unsigned short f2bf(float f) {
  unsigned u = __float_as_uint(f);
  u += 0x7FFFu + ((u >> 16) & 1u);   // round-to-nearest-even
  return (unsigned short)(u >> 16);
}

// ---------------- Pass 1: deterministic global mean / rstd --------------------
__global__ __launch_bounds__(256) void k_reduce_partial(const float* __restrict__ x,
                                                        float* __restrict__ partial) {
  __shared__ float rs[256], rq[256];
  const int64_t base = (int64_t)blockIdx.x * 8192;
  float s = 0.f, q = 0.f;
  for (int k = threadIdx.x; k < 8192; k += 256) {
    float v = x[base + k];
    s += v; q += v * v;
  }
  rs[threadIdx.x] = s; rq[threadIdx.x] = q;
  __syncthreads();
  for (int off = 128; off > 0; off >>= 1) {
    if ((int)threadIdx.x < off) {
      rs[threadIdx.x] += rs[threadIdx.x + off];
      rq[threadIdx.x] += rq[threadIdx.x + off];
    }
    __syncthreads();
  }
  if (threadIdx.x == 0) { partial[blockIdx.x] = rs[0]; partial[1024 + blockIdx.x] = rq[0]; }
}

__global__ __launch_bounds__(256) void k_reduce_final(const float* __restrict__ partial,
                                                      float* __restrict__ stats) {
  __shared__ float rs[256], rq[256];
  float s = 0.f, q = 0.f;
  for (int k = threadIdx.x; k < 1024; k += 256) { s += partial[k]; q += partial[1024 + k]; }
  rs[threadIdx.x] = s; rq[threadIdx.x] = q;
  __syncthreads();
  for (int off = 128; off > 0; off >>= 1) {
    if ((int)threadIdx.x < off) {
      rs[threadIdx.x] += rs[threadIdx.x + off];
      rq[threadIdx.x] += rq[threadIdx.x + off];
    }
    __syncthreads();
  }
  if (threadIdx.x == 0) {
    float mean = rs[0] / (float)NTOT;
    float var  = rq[0] / (float)NTOT - mean * mean;
    stats[0] = mean;
    stats[1] = rsqrtf(var + 1e-5f);
  }
}

// ---- Pass 2: LayerNorm + a/b projections, write bf16 transposed (s-contig) ---
__global__ __launch_bounds__(256) void k_proj(const float* __restrict__ msa,
                                              const float* __restrict__ lnw,
                                              const float* __restrict__ lnb,
                                              const float* __restrict__ wa_w,
                                              const float* __restrict__ wa_b,
                                              const float* __restrict__ wb_w,
                                              const float* __restrict__ wb_b,
                                              const float* __restrict__ stats,
                                              unsigned short* __restrict__ a_t,
                                              unsigned short* __restrict__ b_t) {
  __shared__ float sW[2 * CSM * CM];  // 64 KB: [0..8191]=Wa, [8192..16383]=Wb
  for (int k = threadIdx.x; k < CSM * CM; k += 256) {
    sW[k]             = wa_w[k];
    sW[CSM * CM + k]  = wb_w[k];
  }
  __syncthreads();
  const int s = blockIdx.x;   // 0..127
  const int i = threadIdx.x;  // 0..255
  const float mean = stats[0], rstd = stats[1];
  float accA[CSM], accB[CSM];
#pragma unroll
  for (int c = 0; c < CSM; ++c) { accA[c] = 0.f; accB[c] = 0.f; }
  for (int m = 0; m < CM; ++m) {
    const size_t idx = ((size_t)m * HM + s) * WM + i;
    const float xv = (msa[idx] - mean) * rstd * lnw[idx] + lnb[idx];
#pragma unroll
    for (int c = 0; c < CSM; ++c) {
      accA[c] = fmaf(xv, sW[c * CM + m], accA[c]);
      accB[c] = fmaf(xv, sW[CSM * CM + c * CM + m], accB[c]);
    }
  }
#pragma unroll
  for (int c = 0; c < CSM; ++c) {
    a_t[((size_t)c * WM + i) * HM + s] = f2bf(accA[c] + wa_b[c]);
    b_t[((size_t)c * WM + i) * HM + s] = f2bf(accB[c] + wb_b[c]);
  }
}

// ---------------- wz -> bf16 ---------------------------------------------------
__global__ __launch_bounds__(256) void k_cvt_wz(const float* __restrict__ w,
                                                unsigned short* __restrict__ o, int n) {
  int idx = blockIdx.x * 256 + threadIdx.x;
  if (idx < n) o[idx] = f2bf(w[idx]);
}

// ---- Pass 3: fused outer-product-mean + wz projection, all WMMA --------------
// One workgroup (8 waves) per 16x16 (i,j) tile.
// GEMM1: O_cv[16x16] = A_c^T B_v (K = s = 128), staged bf16 in LDS [n=ij][k=v].
// GEMM2: Z[128 x 256] += Wz[:, c*32+v] @ O, accumulated in VGPRs across c.
__global__ __launch_bounds__(256) void k_opm(const unsigned short* __restrict__ a_t,
                                             const unsigned short* __restrict__ b_t,
                                             const unsigned short* __restrict__ wz_bf,
                                             const float* __restrict__ wz_bias,
                                             float* __restrict__ z_out) {
  __shared__ __align__(16) unsigned short o_lds[256 * 32];  // 16 KB
  const int I0   = blockIdx.x * 16;
  const int J0   = blockIdx.y * 16;
  const int wave = threadIdx.x >> 5;
  const int lane = threadIdx.x & 31;
  const int lm   = lane & 15;
  const int lh   = lane >> 4;

  union Frag { v16bf v; uint4 u[2]; };

  v8f acc[16];
#pragma unroll
  for (int t = 0; t < 16; ++t) acc[t] = (v8f){};

  for (int c = 0; c < CSM; ++c) {
    // ---- GEMM1: this wave computes O for v = wave*4 .. wave*4+3 ----
    Frag afr[4];  // A fragments for each s-chunk, shared across the 4 v's
#pragma unroll
    for (int sc = 0; sc < 4; ++sc) {
      const unsigned short* ap = a_t + ((size_t)(c * WM + I0 + lm) * HM + sc * 32 + lh * 8);
      afr[sc].u[0] = *(const uint4*)ap;
      afr[sc].u[1] = *(const uint4*)(ap + 16);
    }
    unsigned short obf[4][8];
#pragma unroll
    for (int vv = 0; vv < 4; ++vv) {
      const int v = wave * 4 + vv;
      v8f o = (v8f){};
#pragma unroll
      for (int sc = 0; sc < 4; ++sc) {
        Frag bfr;
        const unsigned short* bp = b_t + ((size_t)(v * WM + J0 + lm) * HM + sc * 32 + lh * 8);
        bfr.u[0] = *(const uint4*)bp;
        bfr.u[1] = *(const uint4*)(bp + 16);
        o = __builtin_amdgcn_wmma_f32_16x16x32_bf16(false, afr[sc].v, false, bfr.v,
                                                    (short)0, o, false, false);
      }
#pragma unroll
      for (int r = 0; r < 8; ++r) obf[vv][r] = f2bf(o[r]);
    }

    __syncthreads();  // previous iteration's GEMM2 LDS reads are done
#pragma unroll
    for (int vv = 0; vv < 4; ++vv) {
      const int v = wave * 4 + vv;
#pragma unroll
      for (int r = 0; r < 8; ++r) {
        const int i_loc = lh * 8 + r;   // D layout: lanes16-31 hold M=8..15
        const int j_loc = lm;
        o_lds[(i_loc * 16 + j_loc) * 32 + v] = obf[vv][r];
      }
    }
    __syncthreads();

    // ---- GEMM2: z-block = wave, K-chunk = cv in [c*32, c*32+32) ----
    Frag a2;
    const unsigned short* wp = wz_bf + ((size_t)(wave * 16 + lm) * 1024 + c * 32 + lh * 8);
    a2.u[0] = *(const uint4*)wp;
    a2.u[1] = *(const uint4*)(wp + 16);
#pragma unroll
    for (int nb = 0; nb < 16; ++nb) {
      Frag b2;
      const unsigned short* op = &o_lds[(nb * 16 + lm) * 32 + lh * 8];
      b2.u[0] = *(const uint4*)op;
      b2.u[1] = *(const uint4*)(op + 16);
      acc[nb] = __builtin_amdgcn_wmma_f32_16x16x32_bf16(false, a2.v, false, b2.v,
                                                        (short)0, acc[nb], false, false);
    }
  }

  // ---- store Z: z[z, i, j], apply 1/s and bias ----
  const float inv_s = 1.0f / (float)HM;
#pragma unroll
  for (int nb = 0; nb < 16; ++nb) {
    const int i = I0 + nb;       // n = nb*16 + lm  ->  i_loc = nb, j_loc = lm
    const int j = J0 + lm;
#pragma unroll
    for (int r = 0; r < 8; ++r) {
      const int z = wave * 16 + lh * 8 + r;
      z_out[(size_t)z * (WM * WM) + (size_t)i * WM + j] = acc[nb][r] * inv_s + wz_bias[z];
    }
  }
}

extern "C" void kernel_launch(void* const* d_in, const int* in_sizes, int n_in,
                              void* d_out, int out_size, void* d_ws, size_t ws_size,
                              hipStream_t stream) {
  const float* msa  = (const float*)d_in[0];
  const float* lnw  = (const float*)d_in[1];
  const float* lnb  = (const float*)d_in[2];
  const float* wa_w = (const float*)d_in[3];
  const float* wa_b = (const float*)d_in[4];
  const float* wb_w = (const float*)d_in[5];
  const float* wb_b = (const float*)d_in[6];
  const float* wz_w = (const float*)d_in[7];
  const float* wz_b = (const float*)d_in[8];
  float* z = (float*)d_out;

  char* ws = (char*)d_ws;
  float* stats   = (float*)ws;               // 2 floats
  float* partial = (float*)(ws + 256);       // 2048 floats
  unsigned short* a_t   = (unsigned short*)(ws + 16384);                 // 2 MB
  unsigned short* b_t   = (unsigned short*)(ws + 16384 + 2097152);       // 2 MB
  unsigned short* wz_bf = (unsigned short*)(ws + 16384 + 2 * 2097152);   // 256 KB

  k_reduce_partial<<<1024, 256, 0, stream>>>(msa, partial);
  k_reduce_final<<<1, 256, 0, stream>>>(partial, stats);
  k_proj<<<HM, 256, 0, stream>>>(msa, lnw, lnb, wa_w, wa_b, wb_w, wb_b, stats, a_t, b_t);
  k_cvt_wz<<<(CZ * 1024 + 255) / 256, 256, 0, stream>>>(wz_w, wz_bf, CZ * 1024);

  dim3 grid(WM / 16, WM / 16);
  k_opm<<<grid, 256, 0, stream>>>(a_t, b_t, wz_bf, wz_b, z);
}
